// EWSSM_1675037246122
// MI455X (gfx1250) — matrix operand order repaired and checked by reading, not compiled, not run on hardware
//
#include <hip/hip_runtime.h>
#include <cmath>

typedef __attribute__((ext_vector_type(16))) _Float16 v16h;
typedef __attribute__((ext_vector_type(8)))  float    v8f;

#define N_NODES 20000
#define DM 128
#define DI 256
#define DS 16
#define DTR 8
#define DCONV 4
#define NE 640000
#define EPSV 1e-5f
// chunked scan: 100 chunks x 200 steps (exact for N=20000)
#define CL 200
#define NCH 100

__device__ __forceinline__ float sigmoidf_(float x){ return 1.f/(1.f+__expf(-x)); }
__device__ __forceinline__ float siluf_(float x){ return x*sigmoidf_(x); }
__device__ __forceinline__ float softplusf_(float x){ return (x>20.f)?x:log1pf(__expf(x)); }

// ---- WMMA A-tile loader: 16x32 f16 A from row-major f32 source ----
// ISA layout: lane<16 row m=lane holds k0+0..7 (elems 0-7), k0+16..23 (elems 8-15)
//             lane>=16 row m=lane-16 holds k0+8..15, k0+24..31
__device__ __forceinline__ v16h load_a_tile(const float* __restrict__ src, int ld,
                                            int r0, int k0, int lane){
  const float* p = src + (size_t)(r0 + (lane & 15))*ld + k0 + ((lane < 16) ? 0 : 8);
  v16h a;
#pragma unroll
  for(int i=0;i<8;i++) a[i]   = (_Float16)p[i];
#pragma unroll
  for(int i=0;i<8;i++) a[8+i] = (_Float16)p[16+i];
  return a;
}

// Same, but with fused batchnorm normalize (scale/bias per-k)
__device__ __forceinline__ v16h load_a_tile_bn(const float* __restrict__ src, int ld,
                                               int r0, int k0, int lane,
                                               const float* __restrict__ sc,
                                               const float* __restrict__ bi){
  int kb = k0 + ((lane < 16) ? 0 : 8);
  const float* p = src + (size_t)(r0 + (lane & 15))*ld + kb;
  v16h a;
#pragma unroll
  for(int i=0;i<8;i++) a[i]   = (_Float16)(p[i]   *sc[kb+i]    + bi[kb+i]);
#pragma unroll
  for(int i=0;i<8;i++) a[8+i] = (_Float16)(p[16+i]*sc[kb+16+i] + bi[kb+16+i]);
  return a;
}

// ---- pack weight matrix W[outdim][K] (row-major) into WMMA B lane layout ----
// dst[((nt*kchunks+kc)*32+lane)*16+j] ; h=lane/16: j<8: k=kc*32+8h+j, j>=8: k=kc*32+16+8h+(j-8); col=nt*16+(lane&15)
__global__ void pack_b_kernel(const float* __restrict__ W, _Float16* __restrict__ dst,
                              int K, int kchunks, int total){
  int gid = blockIdx.x*blockDim.x + threadIdx.x;
  if(gid >= total) return;
  int j    = gid & 15;
  int lane = (gid >> 4) & 31;
  int rest = gid >> 9;
  int kc   = rest % kchunks;
  int nt   = rest / kchunks;
  int col  = nt*16 + (lane & 15);
  int k    = kc*32 + ((lane < 16) ? 0 : 8) + ((j < 8) ? j : 8 + j);
  dst[gid] = (_Float16)W[(size_t)col*K + k];
}

// ---- batchnorm stats over rows of (rows x 128) row-major ----
__global__ void bn_stats_kernel(const float* __restrict__ src, int rows,
                                float* __restrict__ sum, float* __restrict__ sq){
  int col = threadIdx.x;                       // 128 threads
  float s = 0.f, s2 = 0.f;
  for(int r = blockIdx.x; r < rows; r += gridDim.x){
    float v = src[(size_t)r*DM + col];
    s += v; s2 += v*v;
  }
  atomicAdd(&sum[col], s);
  atomicAdd(&sq[col], s2);
}

__global__ void bn_finalize_kernel(const float* __restrict__ sum, const float* __restrict__ sq,
                                   float rows, const float* __restrict__ gamma,
                                   const float* __restrict__ beta,
                                   float* __restrict__ scale, float* __restrict__ bias){
  int c = threadIdx.x;
  float m  = sum[c]/rows;
  float v  = sq[c]/rows - m*m;
  float sc = gamma[c]*rsqrtf(v + EPSV);
  scale[c] = sc;
  bias[c]  = beta[c] - m*sc;
}

// ---- fused edge MLP: bn-normalize -> (16 edges x 32 hidden) WMMA -> silu -> dot w2 -> 2x sigmoid
__global__ void edge_mlp_kernel(const float* __restrict__ ea,
                                const float* __restrict__ sc, const float* __restrict__ bi,
                                const _Float16* __restrict__ w1p,
                                const float* __restrict__ b1,
                                const float* __restrict__ w2, const float* __restrict__ b2p,
                                float* __restrict__ ew){
  int wave = threadIdx.x >> 5;
  int lane = threadIdx.x & 31;
  int tile = blockIdx.x*8 + wave;
  if(tile >= NE/16) return;
  int e0 = tile*16;
  v8f c0 = {}; v8f c1 = {};
#pragma unroll
  for(int kc=0;kc<4;kc++){
    v16h a  = load_a_tile_bn(ea, DM, e0, kc*32, lane, sc, bi);
    v16h b0 = *(const v16h*)(w1p + ((size_t)(0*4+kc)*32 + lane)*16);
    v16h b1t= *(const v16h*)(w1p + ((size_t)(1*4+kc)*32 + lane)*16);
    c0 = __builtin_amdgcn_wmma_f32_16x16x32_f16(false, a, false, b0,  (short)0, c0, false, false);
    c1 = __builtin_amdgcn_wmma_f32_16x16x32_f16(false, a, false, b1t, (short)0, c1, false, false);
  }
  // c0: hidden (lane&15), c1: hidden 16+(lane&15); rows = edges (lane<16: m=v, else m=8+v)
  int h0 = lane & 15, h1 = 16 + h0;
  float w2a = w2[h0], w2b = w2[h1];
  float b1a = b1[h0], b1b = b1[h1];
  float b2  = b2p[0];
  float p[8];
#pragma unroll
  for(int v=0; v<8; v++)
    p[v] = siluf_(c0[v]+b1a)*w2a + siluf_(c1[v]+b1b)*w2b;
#pragma unroll
  for(int o=8;o>=1;o>>=1){
#pragma unroll
    for(int v=0;v<8;v++) p[v] += __shfl_xor(p[v], o, 16);
  }
  if((lane & 15) == 0){
    int mb = (lane < 16) ? 0 : 8;
#pragma unroll
    for(int v=0;v<8;v++)
      ew[e0+mb+v] = sigmoidf_(sigmoidf_(p[v] + b2));   // reference applies sigmoid twice
  }
}

// ---- in_proj: xz[ch][node] = sum_k in_proj_w[ch][k]*x[node][k] ----
// register-blocked: one wave = one node tile x 8 output-channel tiles (A loaded once)
__global__ void inproj_kernel(const float* __restrict__ x, const _Float16* __restrict__ wp,
                              float* __restrict__ xz){
  int wave = threadIdx.x >> 5, lane = threadIdx.x & 31;
  int tile = blockIdx.x*8 + wave;               // (N/16) m-tiles * 4 n-groups
  if(tile >= (N_NODES/16)*4) return;
  int mt = tile >> 2, ng = tile & 3;
  int m0 = mt*16;
  v16h a[4];
#pragma unroll
  for(int kc=0;kc<4;kc++) a[kc] = load_a_tile(x, DM, m0, kc*32, lane);
  int nb = m0 + ((lane < 16) ? 0 : 8);
#pragma unroll
  for(int i=0;i<8;i++){
    int nt = ng*8 + i;
    v8f c = {};
#pragma unroll
    for(int kc=0;kc<4;kc++){
      v16h b = *(const v16h*)(wp + ((size_t)(nt*4+kc)*32 + lane)*16);
      c = __builtin_amdgcn_wmma_f32_16x16x32_f16(false, a[kc], false, b, (short)0, c, false, false);
    }
    int ch = nt*16 + (lane & 15);
    float* dst = xz + (size_t)ch*N_NODES + nb;
#pragma unroll
    for(int v=0;v<8;v++) dst[v] = c[v];
  }
}

// ---- causal depthwise conv (k=4) + silu ----
__global__ void conv_silu_kernel(const float* __restrict__ xz, const float* __restrict__ cw,
                                 const float* __restrict__ cb, float* __restrict__ xic){
  int gid = blockIdx.x*blockDim.x + threadIdx.x;
  if(gid >= DI*N_NODES) return;
  int ch = gid / N_NODES;
  int t  = gid % N_NODES;
  const float* row = xz + (size_t)ch*N_NODES;
  float acc = cb[ch];
#pragma unroll
  for(int j=0;j<DCONV;j++){
    int tt = t - (DCONV-1) + j;
    if(tt >= 0) acc += cw[ch*DCONV+j]*row[tt];
  }
  xic[gid] = siluf_(acc);
}

// ---- x_proj: xdbl[r][node], r in 0..39 ----
__global__ void xproj_kernel(const float* __restrict__ xpw, const float* __restrict__ xic,
                             float* __restrict__ xdbl){
  int t = blockIdx.x*blockDim.x + threadIdx.x;
  int r = blockIdx.y;
  if(t >= N_NODES) return;
  const float* wr = xpw + (size_t)r*DI;
  float acc = 0.f;
  for(int d=0; d<DI; d++) acc += wr[d]*xic[(size_t)d*N_NODES + t];
  xdbl[(size_t)r*N_NODES + t] = acc;
}

// ---- delta = softplus(dt_proj_w @ xdbl[:8] + dt_proj_b) ----
__global__ void delta_kernel(const float* __restrict__ dtw, const float* __restrict__ dtb,
                             const float* __restrict__ xdbl, float* __restrict__ delta){
  int gid = blockIdx.x*blockDim.x + threadIdx.x;
  if(gid >= DI*N_NODES) return;
  int ch = gid / N_NODES;
  int t  = gid % N_NODES;
  float acc = dtb[ch];
#pragma unroll
  for(int r=0;r<DTR;r++) acc += dtw[ch*DTR + r]*xdbl[(size_t)r*N_NODES + t];
  delta[gid] = softplusf_(acc);
}

// ---- edge scatter: Bacc[n][dst] += w_e * B0[n][src] (and C) ----
__global__ void edge_agg_kernel(const int* __restrict__ ei, const float* __restrict__ ew,
                                const float* __restrict__ xdbl,
                                float* __restrict__ Bacc, float* __restrict__ Cacc){
  long long gid = (long long)blockIdx.x*blockDim.x + threadIdx.x;
  if(gid >= (long long)NE*DS) return;
  int e = (int)(gid >> 4);
  int n = (int)(gid & 15);
  int src = ei[e], dst = ei[NE + e];
  float w = ew[e];
  atomicAdd(&Bacc[(size_t)n*N_NODES + dst], w * xdbl[(size_t)(DTR+n)*N_NODES + src]);
  atomicAdd(&Cacc[(size_t)n*N_NODES + dst], w * xdbl[(size_t)(DTR+DS+n)*N_NODES + src]);
}

__global__ void aneg_kernel(const float* __restrict__ alog, float* __restrict__ an){
  int i = blockIdx.x*blockDim.x + threadIdx.x;
  if(i < DI*DS) an[i] = -__expf(alog[i]);
}

// ---- chunked associative scan, phase 1: per-chunk (prod dA, local state) ----
__global__ void scan_phase1_kernel(const float* __restrict__ delta, const float* __restrict__ an,
                                   const float* __restrict__ xdbl, const float* __restrict__ Bacc,
                                   const float* __restrict__ xic,
                                   float* __restrict__ Aprod, float* __restrict__ Bloc){
  int c = blockIdx.x;
  int d = blockIdx.y*16 + (threadIdx.x >> 4);
  int n = threadIdx.x & 15;
  float A = an[d*DS + n];
  float ap = 1.f, b = 0.f;
  const float* drow  = delta + (size_t)d*N_NODES;
  const float* xrow  = xic   + (size_t)d*N_NODES;
  const float* b0row = xdbl  + (size_t)(DTR+n)*N_NODES;
  const float* barow = Bacc  + (size_t)n*N_NODES;
  int t0 = c*CL;
  for(int t=t0; t<t0+CL; t++){
    float dl = drow[t];
    float dA = __expf(dl*A);
    b  = dA*b + dl*(b0row[t]+barow[t])*xrow[t];
    ap *= dA;
  }
  int idx = c*4096 + d*16 + n;
  Aprod[idx] = ap; Bloc[idx] = b;
}

// ---- phase 2: serial carry across 100 chunks per (d,n) ----
__global__ void scan_phase2_kernel(const float* __restrict__ Aprod, const float* __restrict__ Bloc,
                                   float* __restrict__ carry){
  int idx = blockIdx.x*blockDim.x + threadIdx.x;   // 4096 threads
  float h = 0.f;
  for(int c=0;c<NCH;c++){
    carry[c*4096 + idx] = h;
    h = Aprod[c*4096 + idx]*h + Bloc[c*4096 + idx];
  }
}

// ---- phase 3: replay with carry, reduce over states, fuse D*xi + silu(z) gate ----
__global__ void scan_phase3_kernel(const float* __restrict__ delta, const float* __restrict__ an,
                                   const float* __restrict__ xdbl, const float* __restrict__ Bacc,
                                   const float* __restrict__ Cacc, const float* __restrict__ xic,
                                   const float* __restrict__ xz, const float* __restrict__ Dp,
                                   const float* __restrict__ carry, float* __restrict__ yg){
  int c = blockIdx.x;
  int d = blockIdx.y*16 + (threadIdx.x >> 4);
  int n = threadIdx.x & 15;
  float A  = an[d*DS + n];
  float h  = carry[c*4096 + d*16 + n];
  float Dv = Dp[d];
  const float* drow  = delta + (size_t)d*N_NODES;
  const float* xrow  = xic   + (size_t)d*N_NODES;
  const float* zrow  = xz    + (size_t)(DI+d)*N_NODES;
  const float* b0row = xdbl  + (size_t)(DTR+n)*N_NODES;
  const float* barow = Bacc  + (size_t)n*N_NODES;
  const float* c0row = xdbl  + (size_t)(DTR+DS+n)*N_NODES;
  const float* carow = Cacc  + (size_t)n*N_NODES;
  int t0 = c*CL;
  for(int t=t0; t<t0+CL; t++){
    float dl = drow[t];
    float xv = xrow[t];
    float dA = __expf(dl*A);
    h = dA*h + dl*(b0row[t]+barow[t])*xv;
    float p = h*(c0row[t]+carow[t]);
#pragma unroll
    for(int o=8;o>=1;o>>=1) p += __shfl_xor(p, o, 16);
    if(n == 0)
      yg[(size_t)t*DI + d] = (p + Dv*xv) * siluf_(zrow[t]);
  }
}

// ---- out_proj: out_pre[node][j] = sum_d yg[node][d]*out_proj_w[j][d] ----
// register-blocked: one wave = one node tile x all 8 output tiles (yg read exactly once)
__global__ void outproj_kernel(const float* __restrict__ yg, const _Float16* __restrict__ wp,
                               float* __restrict__ outp){
  int wave = threadIdx.x >> 5, lane = threadIdx.x & 31;
  int tile = blockIdx.x*8 + wave;               // N/16 m-tiles
  if(tile >= N_NODES/16) return;
  int m0 = tile*16;
  v16h a[8];
#pragma unroll
  for(int kc=0;kc<8;kc++) a[kc] = load_a_tile(yg, DI, m0, kc*32, lane);
  int mb = m0 + ((lane < 16) ? 0 : 8);
#pragma unroll
  for(int nt=0;nt<8;nt++){
    v8f c = {};
#pragma unroll
    for(int kc=0;kc<8;kc++){
      v16h b = *(const v16h*)(wp + ((size_t)(nt*8+kc)*32 + lane)*16);
      c = __builtin_amdgcn_wmma_f32_16x16x32_f16(false, a[kc], false, b, (short)0, c, false, false);
    }
    int j = nt*16 + (lane & 15);
#pragma unroll
    for(int v=0;v<8;v++) outp[(size_t)(mb+v)*DM + j] = c[v];
  }
}

// ---- final batchnorm apply + residual ----
__global__ void bn_apply_residual_kernel(const float* __restrict__ x, const float* __restrict__ outp,
                                         const float* __restrict__ sc, const float* __restrict__ bi,
                                         float* __restrict__ out){
  int gid = blockIdx.x*blockDim.x + threadIdx.x;
  if(gid >= N_NODES*DM) return;
  int col = gid & (DM-1);
  out[gid] = x[gid] + outp[gid]*sc[col] + bi[col];
}

extern "C" void kernel_launch(void* const* d_in, const int* in_sizes, int n_in,
                              void* d_out, int out_size, void* d_ws, size_t ws_size,
                              hipStream_t stream) {
  (void)in_sizes; (void)n_in; (void)out_size; (void)ws_size;
  const float* x        = (const float*)d_in[0];
  const float* ea       = (const float*)d_in[1];
  const int*   ei       = (const int*)  d_in[2];
  const float* ew_gamma = (const float*)d_in[3];
  const float* ew_beta  = (const float*)d_in[4];
  const float* ew_w1    = (const float*)d_in[5];
  const float* ew_b1    = (const float*)d_in[6];
  const float* ew_w2    = (const float*)d_in[7];
  const float* ew_b2    = (const float*)d_in[8];
  const float* in_proj_w= (const float*)d_in[9];
  const float* conv_w   = (const float*)d_in[10];
  const float* conv_b   = (const float*)d_in[11];
  const float* x_proj_w = (const float*)d_in[12];
  const float* dt_proj_w= (const float*)d_in[13];
  const float* dt_proj_b= (const float*)d_in[14];
  const float* A_log    = (const float*)d_in[15];
  const float* D_param  = (const float*)d_in[16];
  const float* out_proj_w=(const float*)d_in[17];
  const float* out_gamma= (const float*)d_in[18];
  const float* out_beta = (const float*)d_in[19];
  float* out = (float*)d_out;

  // bump allocator over workspace
  char* base = (char*)d_ws;
  size_t off = 0;
  auto alloc = [&](size_t bytes) -> char* {
    char* p = base + off;
    off += (bytes + 255) & ~(size_t)255;
    return p;
  };
  float*    esum   = (float*)alloc(DM*4);
  float*    esq    = (float*)alloc(DM*4);
  float*    escale = (float*)alloc(DM*4);
  float*    ebias  = (float*)alloc(DM*4);
  float*    osum   = (float*)alloc(DM*4);
  float*    osq    = (float*)alloc(DM*4);
  float*    oscale = (float*)alloc(DM*4);
  float*    obias  = (float*)alloc(DM*4);
  _Float16* w1p    = (_Float16*)alloc((size_t)2*4*32*16*2);
  _Float16* inwp   = (_Float16*)alloc((size_t)32*4*32*16*2);
  _Float16* outwp  = (_Float16*)alloc((size_t)8*8*32*16*2);
  float*    ew     = (float*)alloc((size_t)NE*4);
  float*    xz     = (float*)alloc((size_t)2*DI*N_NODES*4);
  float*    xic    = (float*)alloc((size_t)DI*N_NODES*4);
  float*    xdbl   = (float*)alloc((size_t)(DTR+2*DS)*N_NODES*4);
  float*    delta  = (float*)alloc((size_t)DI*N_NODES*4);
  float*    Bacc   = (float*)alloc((size_t)DS*N_NODES*4);
  float*    Cacc   = (float*)alloc((size_t)DS*N_NODES*4);
  float*    an     = (float*)alloc((size_t)DI*DS*4);
  float*    Aprod  = (float*)alloc((size_t)NCH*4096*4);
  float*    Bloc   = (float*)alloc((size_t)NCH*4096*4);
  float*    carry  = (float*)alloc((size_t)NCH*4096*4);
  float*    yg     = (float*)alloc((size_t)N_NODES*DI*4);
  float*    outp   = (float*)alloc((size_t)N_NODES*DM*4);

  // zero atomics targets (graph-capturable)
  hipMemsetAsync(esum, 0, DM*4, stream);
  hipMemsetAsync(esq,  0, DM*4, stream);
  hipMemsetAsync(osum, 0, DM*4, stream);
  hipMemsetAsync(osq,  0, DM*4, stream);
  hipMemsetAsync(Bacc, 0, (size_t)DS*N_NODES*4, stream);
  hipMemsetAsync(Cacc, 0, (size_t)DS*N_NODES*4, stream);

  // weight packing into WMMA B layout
  pack_b_kernel<<<(2*4*32*16+255)/256, 256, 0, stream>>>(ew_w1,     w1p,  DM, 4, 2*4*32*16);
  pack_b_kernel<<<(32*4*32*16+255)/256,256, 0, stream>>>(in_proj_w, inwp, DM, 4, 32*4*32*16);
  pack_b_kernel<<<(8*8*32*16+255)/256, 256, 0, stream>>>(out_proj_w,outwp, DI, 8, 8*8*32*16);

  // edge batchnorm stats -> scale/bias
  bn_stats_kernel<<<1024, DM, 0, stream>>>(ea, NE, esum, esq);
  bn_finalize_kernel<<<1, DM, 0, stream>>>(esum, esq, (float)NE, ew_gamma, ew_beta, escale, ebias);

  // fused edge MLP (WMMA)
  edge_mlp_kernel<<<(NE/16 + 7)/8, 256, 0, stream>>>(ea, escale, ebias, w1p, ew_b1, ew_w2, ew_b2, ew);

  // in_proj GEMM (WMMA, register-blocked 1x8 tiles)
  inproj_kernel<<<((N_NODES/16)*4 + 7)/8, 256, 0, stream>>>(x, inwp, xz);

  // conv + silu
  conv_silu_kernel<<<(DI*N_NODES + 255)/256, 256, 0, stream>>>(xz, conv_w, conv_b, xic);

  // x_proj and delta
  {
    dim3 g((N_NODES + 255)/256, DTR + 2*DS);
    xproj_kernel<<<g, 256, 0, stream>>>(x_proj_w, xic, xdbl);
  }
  delta_kernel<<<(DI*N_NODES + 255)/256, 256, 0, stream>>>(dt_proj_w, dt_proj_b, xdbl, delta);

  // edge-weighted scatter into Bacc/Cacc
  edge_agg_kernel<<<(int)(((long long)NE*DS + 255)/256), 256, 0, stream>>>(ei, ew, xdbl, Bacc, Cacc);

  // A = -exp(A_log)
  aneg_kernel<<<(DI*DS + 255)/256, 256, 0, stream>>>(A_log, an);

  // chunked scan
  scan_phase1_kernel<<<dim3(NCH, 16), 256, 0, stream>>>(delta, an, xdbl, Bacc, xic, Aprod, Bloc);
  scan_phase2_kernel<<<16, 256, 0, stream>>>(Aprod, Bloc, carry);
  scan_phase3_kernel<<<dim3(NCH, 16), 256, 0, stream>>>(delta, an, xdbl, Bacc, Cacc, xic, xz,
                                                        D_param, carry, yg);

  // out_proj GEMM (WMMA, register-blocked 1x8 tiles, single pass over yg)
  outproj_kernel<<<(N_NODES/16 + 7)/8, 256, 0, stream>>>(yg, outwp, outp);

  // output batchnorm + residual
  bn_stats_kernel<<<1024, DM, 0, stream>>>(outp, N_NODES, osum, osq);
  bn_finalize_kernel<<<1, DM, 0, stream>>>(osum, osq, (float)N_NODES, out_gamma, out_beta, oscale, obias);
  bn_apply_residual_kernel<<<(N_NODES*DM + 255)/256, 256, 0, stream>>>(x, outp, oscale, obias, out);
}
